// PatchedPhiMoESparseMoeBlock_59055800320749
// MI455X (gfx1250) — matrix operand order repaired and checked by reading, not compiled, not run on hardware
//
#include <hip/hip_runtime.h>
#include <hip/hip_bf16.h>
#include <math.h>

#define T_TOKENS 8192
#define HID      1024
#define FFN      2048
#define E_NUM    16
#define JEPS     0.01f

typedef __attribute__((ext_vector_type(16))) __bf16       v16bf;
typedef __attribute__((ext_vector_type(8)))  float        v8f;
typedef __attribute__((ext_vector_type(4)))  unsigned int v4u;

union AFrag { v4u u[2]; v16bf v; };

#define WMMA_BF16(A, B, C) \
  __builtin_amdgcn_wmma_f32_16x16x32_bf16(false, (A), false, (B), (short)0, (C), false, false)

__device__ __forceinline__ unsigned short f2bf(float f) {
  unsigned int u = __float_as_uint(f);
  u += 0x7FFFu + ((u >> 16) & 1u);   // round-to-nearest-even
  return (unsigned short)(u >> 16);
}

// ---------------------------------------------------------------- zeroing
__global__ void zero_f32_kernel(float* __restrict__ p, size_t n) {
  size_t i = (size_t)blockIdx.x * blockDim.x + threadIdx.x;
  size_t stride = (size_t)gridDim.x * blockDim.x;
  for (; i < n; i += stride) p[i] = 0.f;
}

__global__ void zero_i32_kernel(int* __restrict__ p, int n) {
  int i = blockIdx.x * blockDim.x + threadIdx.x;
  if (i < n) p[i] = 0;
}

// ---------------------------------------------------------------- fp32 -> bf16
__global__ void cvt_bf16_kernel(const float* __restrict__ src,
                                unsigned short* __restrict__ dst, size_t n) {
  size_t i = ((size_t)blockIdx.x * blockDim.x + threadIdx.x) * 2;
  size_t stride = (size_t)gridDim.x * blockDim.x * 2;
  for (; i < n; i += stride) {
    float2 f = *(const float2*)(src + i);
    unsigned int lo = f2bf(f.x);
    unsigned int hi = f2bf(f.y);
    *(unsigned int*)(dst + i) = lo | (hi << 16);
  }
}

// ---------------------------------------------------------------- router logits
__global__ void router_kernel(const float* __restrict__ x,
                              const float* __restrict__ gw,
                              float* __restrict__ logits) {
  int gid = blockIdx.x * blockDim.x + threadIdx.x;   // T*E threads
  int t = gid >> 4, e = gid & 15;
  const float4* xr = (const float4*)(x + (size_t)t * HID);
  const float4* gr = (const float4*)(gw + (size_t)e * HID);
  float acc = 0.f;
#pragma unroll 8
  for (int k = 0; k < HID / 4; ++k) {
    float4 a = xr[k], b = gr[k];
    acc += a.x * b.x + a.y * b.y + a.z * b.z + a.w * b.w;
  }
  logits[(size_t)t * E_NUM + e] = acc;
}

// ---------------------------------------------------------------- sparsemixer + bucketize
__global__ void route_kernel(const float* __restrict__ logits,
                             int* __restrict__ counts,
                             int* __restrict__ etok,
                             float* __restrict__ ewt) {
  int t = blockIdx.x * blockDim.x + threadIdx.x;
  if (t >= T_TOKENS) return;
  float s[E_NUM];
#pragma unroll
  for (int i = 0; i < E_NUM; ++i) s[i] = logits[(size_t)t * E_NUM + i];

  // top-1 (first occurrence of max)
  float m1 = s[0]; int i1 = 0;
#pragma unroll
  for (int i = 1; i < E_NUM; ++i) if (s[i] > m1) { m1 = s[i]; i1 = i; }
  // softmax with jitter mask; top-1 never masked -> w1 = exp(0)/denom
  float d1 = 0.f;
#pragma unroll
  for (int i = 0; i < E_NUM; ++i) {
    float fac = fmaxf(fabsf(s[i]), m1);
    bool masked = ((m1 - s[i]) / fac) > (2.f * JEPS);
    if (!masked) d1 += __expf(s[i] - m1);
  }
  float w1 = 1.f / d1;

  // top-2 over scores with top-1 masked out
  float m2 = -3.4e38f; int i2 = 0;
#pragma unroll
  for (int i = 0; i < E_NUM; ++i)
    if (i != i1 && s[i] > m2) { m2 = s[i]; i2 = i; }
  float d2 = 0.f;
#pragma unroll
  for (int i = 0; i < E_NUM; ++i) {
    if (i == i1) continue;
    float fac = fmaxf(fabsf(s[i]), m2);
    bool masked = ((m2 - s[i]) / fac) > (2.f * JEPS);
    if (!masked) d2 += __expf(s[i] - m2);
  }
  float w2 = 1.f / d2;

  int s1 = atomicAdd(&counts[i1], 1);
  etok[i1 * T_TOKENS + s1] = t; ewt[i1 * T_TOKENS + s1] = w1;
  int s2 = atomicAdd(&counts[i2], 1);
  etok[i2 * T_TOKENS + s2] = t; ewt[i2 * T_TOKENS + s2] = w2;
}

// ---------------------------------------------------------------- fused expert FFN
// One block = (expert e, 64-token tile), 8 waves.
// Per wave: 4 row-blocks x 4 B-fragments of WMMA accumulators (128 VGPRs).
// Stage 1: gu = x @ Wgu^T (bf16 WMMA), h = silu(gu_lo)*gu_hi -> LDS (64 x 2048 bf16 = 256KB).
// Stage 2: out += w_t * (h @ Wdn^T), A-fragments from LDS, f32 atomics to global.
__global__ __launch_bounds__(256) void moe_kernel(
    const unsigned short* __restrict__ xb,     // [T, H] bf16
    const unsigned short* __restrict__ wgu,    // [E, 2F, H] bf16
    const unsigned short* __restrict__ wdn,    // [E, H, F] bf16
    const int* __restrict__ counts,
    const int* __restrict__ etok,
    const float* __restrict__ ewt,
    float* __restrict__ out)                   // [T, H] f32
{
  __shared__ unsigned short hbuf[64 * FFN];    // 256 KB (CDNA5: 320KB LDS/WGP)
  __shared__ int   idx_s[64];
  __shared__ float wt_s[64];

  int e    = blockIdx.x >> 7;                  // 128 tiles per expert
  int tile = blockIdx.x & 127;
  int cnt  = counts[e];
  int m_base = tile * 64;
  if (m_base >= cnt) return;                   // uniform exit

  int tid = threadIdx.x;
  if (tid < 64) {
    int slot = m_base + tid;
    int tok = 0; float w = 0.f;                // padded slots: token 0, weight 0
    if (slot < cnt) { tok = etok[e * T_TOKENS + slot]; w = ewt[e * T_TOKENS + slot]; }
    idx_s[tid] = tok; wt_s[tid] = w;
  }
  __syncthreads();

  int wv = tid >> 5, l = tid & 31;
  int hi = l >> 4, ln = l & 15;
  const v8f vzero = {0.f,0.f,0.f,0.f,0.f,0.f,0.f,0.f};

  const unsigned short* wgu_e = wgu + (size_t)e * (2 * FFN) * HID;
  const unsigned short* wdn_e = wdn + (size_t)e * HID * FFN;

  // A-fragment row pointers (k order chosen here, same for B: lane ln = row, K contiguous at +16*hi .. +16*hi+15)
  const unsigned short* xrow[4];
#pragma unroll
  for (int r = 0; r < 4; ++r)
    xrow[r] = xb + (size_t)idx_s[16 * r + ln] * HID + 16 * hi;

  // -------- stage 1: gate_up + SiLU into LDS --------
  // n-chunk = 256 dual columns; wave wv owns 32 dual columns (2 tiles of 16, lo+hi each)
  for (int n0 = 0; n0 < FFN; n0 += 256) {
    int nb = n0 + wv * 32;
    const unsigned short* bp[4];               // q: 0=c0 gate, 1=c0 up, 2=c1 gate, 3=c1 up
    bp[0] = wgu_e + (size_t)(nb + ln) * HID + 16 * hi;
    bp[1] = wgu_e + (size_t)(nb + FFN + ln) * HID + 16 * hi;
    bp[2] = wgu_e + (size_t)(nb + 16 + ln) * HID + 16 * hi;
    bp[3] = wgu_e + (size_t)(nb + 16 + FFN + ln) * HID + 16 * hi;

    v8f acc[4][4];
#pragma unroll
    for (int r = 0; r < 4; ++r)
#pragma unroll
      for (int q = 0; q < 4; ++q) acc[r][q] = vzero;

#pragma unroll 2
    for (int k0 = 0; k0 < HID; k0 += 32) {
      AFrag a[4];
#pragma unroll
      for (int r = 0; r < 4; ++r) {
        a[r].u[0] = *(const v4u*)(xrow[r] + k0);
        a[r].u[1] = *(const v4u*)(xrow[r] + k0 + 8);
      }
      v16bf b[4];
#pragma unroll
      for (int q = 0; q < 4; ++q) b[q] = *(const v16bf*)(bp[q] + k0);
#pragma unroll
      for (int r = 0; r < 4; ++r)
#pragma unroll
        for (int q = 0; q < 4; ++q)
          acc[r][q] = WMMA_BF16(a[r].v, b[q], acc[r][q]);
    }

    // SiLU(gate) * up  -> LDS  (C/D layout: row = i + 8*hi, col = ln)
#pragma unroll
    for (int r = 0; r < 4; ++r)
#pragma unroll
      for (int c = 0; c < 2; ++c)
#pragma unroll
        for (int i = 0; i < 8; ++i) {
          float lo = acc[r][2 * c][i], up = acc[r][2 * c + 1][i];
          float hval = (lo / (1.f + __expf(-lo))) * up;
          hbuf[(16 * r + i + 8 * hi) * FFN + nb + 16 * c + ln] = f2bf(hval);
        }
  }
  __syncthreads();

  // -------- stage 2: down projection + weighted scatter --------
  const unsigned short* hrow[4];
#pragma unroll
  for (int r = 0; r < 4; ++r)
    hrow[r] = &hbuf[(16 * r + ln) * FFN + 16 * hi];

#pragma unroll 1
  for (int jo = 0; jo < 2; ++jo) {
    int j0 = wv * 64 + jo * 512;               // wave's 64 output columns
    const unsigned short* bd[4];
#pragma unroll
    for (int c = 0; c < 4; ++c)
      bd[c] = wdn_e + (size_t)(j0 + 16 * c + ln) * FFN + 16 * hi;

    v8f acc[4][4];
#pragma unroll
    for (int r = 0; r < 4; ++r)
#pragma unroll
      for (int c = 0; c < 4; ++c) acc[r][c] = vzero;

#pragma unroll 2
    for (int f0 = 0; f0 < FFN; f0 += 32) {
      AFrag a[4];
#pragma unroll
      for (int r = 0; r < 4; ++r) {
        a[r].u[0] = *(const v4u*)(hrow[r] + f0);        // ds_load_b128
        a[r].u[1] = *(const v4u*)(hrow[r] + f0 + 8);
      }
      v16bf b[4];
#pragma unroll
      for (int c = 0; c < 4; ++c) b[c] = *(const v16bf*)(bd[c] + f0);
#pragma unroll
      for (int r = 0; r < 4; ++r)
#pragma unroll
        for (int c = 0; c < 4; ++c)
          acc[r][c] = WMMA_BF16(a[r].v, b[c], acc[r][c]);
    }

#pragma unroll
    for (int r = 0; r < 4; ++r)
#pragma unroll
      for (int i = 0; i < 8; ++i) {
        int m = 16 * r + i + 8 * hi;
        int tok = idx_s[m];
        float w = wt_s[m];
        float* orow = out + (size_t)tok * HID + j0 + ln;
#pragma unroll
        for (int c = 0; c < 4; ++c)
          atomicAdd(orow + 16 * c, acc[r][c][i] * w);
      }
  }
}

// ---------------------------------------------------------------- launch
extern "C" void kernel_launch(void* const* d_in, const int* in_sizes, int n_in,
                              void* d_out, int out_size, void* d_ws, size_t ws_size,
                              hipStream_t stream) {
  const float* x   = (const float*)d_in[0];   // [4,2048,1024]
  const float* gw  = (const float*)d_in[1];   // [16,1024]
  const float* wgu = (const float*)d_in[2];   // [16,4096,1024]
  const float* wdn = (const float*)d_in[3];   // [16,1024,2048]
  float* out    = (float*)d_out;                         // [T,H] f32
  float* logits = out + (size_t)T_TOKENS * HID;          // [T,E] f32 (output tail)

  char* ws = (char*)d_ws;
  size_t off = 0;
  unsigned short* xb   = (unsigned short*)(ws + off); off += (size_t)T_TOKENS * HID * 2;
  unsigned short* wgub = (unsigned short*)(ws + off); off += (size_t)E_NUM * 2 * FFN * HID * 2;
  unsigned short* wdnb = (unsigned short*)(ws + off); off += (size_t)E_NUM * HID * FFN * 2;
  int*   counts = (int*)(ws + off);   off += 256;
  int*   etok   = (int*)(ws + off);   off += (size_t)E_NUM * T_TOKENS * 4;
  float* ewt    = (float*)(ws + off); off += (size_t)E_NUM * T_TOKENS * 4;
  (void)ws_size; (void)in_sizes; (void)n_in; (void)out_size;

  zero_f32_kernel<<<2048, 256, 0, stream>>>(out, (size_t)T_TOKENS * HID);
  zero_i32_kernel<<<1, 16, 0, stream>>>(counts, E_NUM);

  cvt_bf16_kernel<<<2048, 256, 0, stream>>>(x,   xb,   (size_t)T_TOKENS * HID);
  cvt_bf16_kernel<<<4096, 256, 0, stream>>>(wgu, wgub, (size_t)E_NUM * 2 * FFN * HID);
  cvt_bf16_kernel<<<4096, 256, 0, stream>>>(wdn, wdnb, (size_t)E_NUM * HID * FFN);

  router_kernel<<<(T_TOKENS * E_NUM) / 256, 256, 0, stream>>>(x, gw, logits);
  route_kernel<<<T_TOKENS / 256, 256, 0, stream>>>(logits, counts, etok, ewt);

  moe_kernel<<<E_NUM * 128, 256, 0, stream>>>(xb, wgub, wdnb, counts, etok, ewt, out);
}
